// HookedMambaBlockBatchSplit_57518202028336
// MI455X (gfx1250) — compile-verified, matrix operand order checked by reading, not run on hardware
//
#include <hip/hip_runtime.h>

// ---------------------------------------------------------------------------
// Problem constants (match the reference)
// ---------------------------------------------------------------------------
constexpr int  Bc = 2, Lc = 2048, Dc = 2048, Ec = 4096, Nc = 16, Rc = 128, Kc = 4;
constexpr long BLc = (long)Bc * Lc;   // 4096 rows

// ---------------------------------------------------------------------------
// Types for WMMA / TDM (CDNA5 / gfx1250, wave32)
// ---------------------------------------------------------------------------
typedef __attribute__((ext_vector_type(16))) __bf16 v16bf;
typedef __attribute__((ext_vector_type(8)))  __bf16 v8bf;
typedef __attribute__((ext_vector_type(8)))  float  v8f;
typedef __attribute__((ext_vector_type(4)))  unsigned int v4u;
typedef __attribute__((ext_vector_type(4)))  int v4i;
typedef __attribute__((ext_vector_type(8)))  int v8i;

static __device__ inline __bf16 f2bf(float f) {
  union { float f; unsigned u; } v; v.f = f;
  unsigned r = v.u + 0x7FFFu + ((v.u >> 16) & 1u);   // round-to-nearest-even
  unsigned short h = (unsigned short)(r >> 16);
  return __builtin_bit_cast(__bf16, h);
}

static __device__ inline unsigned lds_addr32(const void* p) {
  // Low 32 bits of a shared-aperture address are the LDS byte offset.
  return (unsigned)(unsigned long long)p;
}

// ---------------------------------------------------------------------------
// TDM: async 2D tile load (rows x 32 bf16) global -> LDS.
//  - row stride in memory: row_stride_elems (bf16 units)
//  - LDS: rows packed at 64B + 16B pad = 80B stride (pad_interval=16 DW,
//    pad_amount=4 DW), keeping ds_load_b128 fragment reads conflict-spread.
//  - rows_remaining < tile_rows => hardware OOB zero-fill (used for N=16/128).
// ---------------------------------------------------------------------------
static __device__ inline void tdm_load_rows32_bf16(const __bf16* gtile, unsigned ldsoff,
                                                   int tile_rows, int rows_remaining,
                                                   long row_stride_elems) {
  const unsigned long long ga = (unsigned long long)(const void*)gtile;
  v4u g0;
  g0[0] = 1u;                                             // count=1 (valid user D#)
  g0[1] = ldsoff;                                         // lds_addr (bytes)
  g0[2] = (unsigned)ga;                                   // global_addr[31:0]
  g0[3] = (unsigned)((ga >> 32) & 0x01FFFFFFull)          // global_addr[56:32]
        | (2u << 30);                                     // type = 2 ("image")

  const unsigned td0 = 32u;                               // tensor_dim0 (K in slab)
  const unsigned td1 = (unsigned)rows_remaining;          // tensor_dim1 (OOB edge)
  const unsigned long long s0 = (unsigned long long)row_stride_elems;
  v8i g1;
  g1[0] = (int)((1u << 16)        // data_size = 2 bytes
              | (1u << 20)        // pad_enable
              | (3u << 22)        // pad_interval: 16 DWORDs (one 64B row)
              | (3u << 25));      // pad_amount:   4 DWORDs (16B)
  g1[1] = (int)((td0 & 0xFFFFu) << 16);                   // tensor_dim0[15:0]
  g1[2] = (int)((td0 >> 16) | ((td1 & 0xFFFFu) << 16));   // td0[31:16] | td1[15:0]
  g1[3] = (int)((td1 >> 16) | (32u << 16));               // td1[31:16] | tile_dim0=32
  g1[4] = (int)((unsigned)tile_rows & 0xFFFFu);           // tile_dim1 (tile_dim2=0)
  g1[5] = (int)(unsigned)(s0 & 0xFFFFFFFFull);            // tensor_dim0_stride[31:0]
  g1[6] = (int)(unsigned)((s0 >> 32) & 0xFFFFull);        // stride[47:32]
  g1[7] = 0;
  const v4i z4 = {0, 0, 0, 0};                            // groups 2/3 unused (2D)
  const v8i z8 = {0, 0, 0, 0, 0, 0, 0, 0};
  __builtin_amdgcn_tensor_load_to_lds(g0, g1, z4, z4, z8, 0);
}

// ---------------------------------------------------------------------------
// Generic fp32 -> bf16 conversion
// ---------------------------------------------------------------------------
__global__ __launch_bounds__(256)
void cvt_f32_bf16_kernel(const float* __restrict__ in, __bf16* __restrict__ out, long n) {
  long i = (long)blockIdx.x * 256 + threadIdx.x;
  if (i < n) out[i] = f2bf(in[i]);
}

// ---------------------------------------------------------------------------
// RMSNorm over D, write bf16 activations
// ---------------------------------------------------------------------------
__global__ __launch_bounds__(256)
void rmsnorm_kernel(const float* __restrict__ resid, const float* __restrict__ nw,
                    __bf16* __restrict__ xn) {
  __shared__ float red[8];
  __shared__ float stot;
  const long row = blockIdx.x;
  const float* r = resid + row * Dc;
  float s = 0.f;
  for (int i = threadIdx.x; i < Dc; i += 256) { float v = r[i]; s += v * v; }
  for (int off = 16; off > 0; off >>= 1) s += __shfl_down(s, off, 32);
  if ((threadIdx.x & 31) == 0) red[threadIdx.x >> 5] = s;
  __syncthreads();
  if (threadIdx.x == 0) {
    float t = 0.f;
    for (int i = 0; i < 8; i++) t += red[i];
    stot = t;
  }
  __syncthreads();
  const float scale = rsqrtf(stot / (float)Dc + 1e-5f);
  for (int i = threadIdx.x; i < Dc; i += 256)
    xn[row * Dc + i] = f2bf(r[i] * scale * nw[i]);
}

// ---------------------------------------------------------------------------
// Tiled bf16 WMMA GEMM with TDM double-buffered LDS staging:
//   C[M,ldc] (f32) = A[M,K] @ W[N,K]^T  (+ optional Cadd)
// 256 threads / 8 waves, 128(M) x 256(N) block, K in 32-wide slabs.
// Wave grid 2(M) x 4(N); each wave owns 4x4 tiles of 16x16 -> 16 WMMA/slab.
// Wave 0 issues TENSOR_LOAD_TO_LDS for slab s+1 while all waves compute s.
// ---------------------------------------------------------------------------
#define GT_M 128
#define GT_N 256
#define GT_K 32
#define ROWE 40                       // bf16 elems per LDS row (32 + 8 pad = 80B)
#define BUF_ELEMS ((GT_M + GT_N) * ROWE)

__global__ __launch_bounds__(256)
void gemm_bf16_wmma(const __bf16* __restrict__ A,    // [M, Kdim] row-major
                    const __bf16* __restrict__ W,    // [Ndim, Kdim] row-major
                    const float*  __restrict__ Cadd, // nullptr or [M, ldc]
                    float* __restrict__ C,           // [M, ldc]
                    int M, int Ndim, int Kdim, int ldc) {
  __shared__ __bf16 lds[2 * BUF_ELEMS];              // 61,440 B

  const int tid  = threadIdx.x;
  const int lane = tid & 31;
  const int wave = tid >> 5;       // 0..7
  const int hf   = lane >> 4;      // K phase (lanes 0-15 vs 16-31)
  const int lr   = lane & 15;

  const int wm = wave & 1;         // 2 waves along M, 64 rows each
  const int wn = wave >> 1;        // 4 waves along N, 64 cols each

  const int m0 = blockIdx.x * GT_M;
  const int n0 = blockIdx.y * GT_N;

  v8f acc[4][4];
#pragma unroll
  for (int i = 0; i < 4; i++)
#pragma unroll
    for (int j = 0; j < 4; j++) acc[i][j] = (v8f)0.0f;

  const int nslab = Kdim / GT_K;
  const long aRow0 = (long)m0;
  const long bRow0 = (long)n0;

  // ---- prologue: slab 0 into buffer 0 ----
  if (wave == 0) {
    tdm_load_rows32_bf16(A + aRow0 * Kdim, lds_addr32(&lds[0]),
                         GT_M, M - m0, Kdim);
    tdm_load_rows32_bf16(W + bRow0 * Kdim, lds_addr32(&lds[GT_M * ROWE]),
                         GT_N, Ndim - n0, Kdim);
    __builtin_amdgcn_s_wait_tensorcnt((short)0);
  }
  __syncthreads();

  int buf = 0;
  for (int s = 0; s < nslab; s++) {
    // ---- kick off slab s+1 into the other buffer (overlaps compute) ----
    if (wave == 0 && (s + 1) < nslab) {
      const int k1 = (s + 1) * GT_K;
      const unsigned obase = (buf ^ 1) * BUF_ELEMS;
      tdm_load_rows32_bf16(A + aRow0 * Kdim + k1, lds_addr32(&lds[obase]),
                           GT_M, M - m0, Kdim);
      tdm_load_rows32_bf16(W + bRow0 * Kdim + k1, lds_addr32(&lds[obase + GT_M * ROWE]),
                           GT_N, Ndim - n0, Kdim);
    }

    // ---- fragments from current buffer ----
    const __bf16* bA = &lds[buf * BUF_ELEMS];
    const __bf16* bB = bA + GT_M * ROWE;

    v16bf afrag[4], bfrag[4];
#pragma unroll
    for (int i = 0; i < 4; i++) {
      const int row = wm * 64 + i * 16 + lr;
      const __bf16* p = bA + row * ROWE + hf * 8;
      v8bf lo = *(const v8bf*)(p);
      v8bf hi = *(const v8bf*)(p + 16);
#pragma unroll
      for (int q = 0; q < 8; q++) { afrag[i][q] = lo[q]; afrag[i][q + 8] = hi[q]; }
    }
#pragma unroll
    for (int j = 0; j < 4; j++) {
      const int row = wn * 64 + j * 16 + lr;
      const __bf16* p = bB + row * ROWE + hf * 8;
      v8bf lo = *(const v8bf*)(p);
      v8bf hi = *(const v8bf*)(p + 16);
#pragma unroll
      for (int q = 0; q < 8; q++) { bfrag[j][q] = lo[q]; bfrag[j][q + 8] = hi[q]; }
    }

    // ---- 16 WMMAs per wave per slab ----
#pragma unroll
    for (int i = 0; i < 4; i++)
#pragma unroll
      for (int j = 0; j < 4; j++)
        acc[i][j] = __builtin_amdgcn_wmma_f32_16x16x32_bf16(
            false, afrag[i], false, bfrag[j], (short)0, acc[i][j], false, false);

    // ---- fence next slab's DMA, publish to all waves ----
    if (wave == 0) __builtin_amdgcn_s_wait_tensorcnt((short)0);
    __syncthreads();
    buf ^= 1;
  }

  // ---- store (C/D layout: lane col = lr, VGPR v -> row v + hf*8) ----
#pragma unroll
  for (int i = 0; i < 4; i++) {
    const int row0 = m0 + wm * 64 + i * 16 + hf * 8;
#pragma unroll
    for (int j = 0; j < 4; j++) {
      const int col = n0 + wn * 64 + j * 16 + lr;
      if (col < Ndim) {
#pragma unroll
        for (int v = 0; v < 8; v++) {
          const long idx = (long)(row0 + v) * ldc + col;
          float val = acc[i][j][v];
          if (Cadd) val += Cadd[idx];
          C[idx] = val;
        }
      }
    }
  }
}

// ---------------------------------------------------------------------------
// Depthwise causal conv (K=4) + SiLU; writes f32 and bf16 copies
// ---------------------------------------------------------------------------
__global__ __launch_bounds__(256)
void conv_silu_kernel(const float* __restrict__ xin, const float* __restrict__ cw,
                      const float* __restrict__ cb, float* __restrict__ xs,
                      __bf16* __restrict__ xs_bf) {
  const long idx = (long)blockIdx.x * 256 + threadIdx.x;  // over BL*E
  const int  e   = (int)(idx % Ec);
  const long bl  = idx / Ec;
  const int  l   = (int)(bl % Lc);
  float acc = cb[e];
#pragma unroll
  for (int k = 0; k < Kc; k++) {
    const int ls = l - (Kc - 1) + k;
    if (ls >= 0) acc += cw[e * Kc + k] * xin[(bl - (Kc - 1) + k) * Ec + e];
  }
  const float sx = acc / (1.f + __expf(-acc));
  xs[idx] = sx;
  xs_bf[idx] = f2bf(sx);
}

// ---------------------------------------------------------------------------
// delta = softplus(delta_pre + b_d2[e])   (in place on delta buffer)
// ---------------------------------------------------------------------------
__global__ __launch_bounds__(256)
void softplus_kernel(float* __restrict__ dlt, const float* __restrict__ bd2) {
  const long idx = (long)blockIdx.x * 256 + threadIdx.x;
  const int  e   = (int)(idx % Ec);
  const float v  = dlt[idx] + bd2[e];
  dlt[idx] = (v > 20.f) ? v : log1pf(__expf(v));
}

// ---------------------------------------------------------------------------
// Selective scan: one thread per channel e, h[16] state in registers,
// Bt/Ct loads are block-uniform (scalarizable).
// ---------------------------------------------------------------------------
__global__ __launch_bounds__(256)
void scan_kernel(const float* __restrict__ dlt, const float* __restrict__ xs,
                 const float* __restrict__ Bt, const float* __restrict__ Ct,
                 const float* __restrict__ A_log, float* __restrict__ y) {
  const int e = blockIdx.x * 256 + threadIdx.x;
  const int b = blockIdx.y;
  float Ar[Nc], h[Nc];
#pragma unroll
  for (int n = 0; n < Nc; n++) { Ar[n] = -__expf(A_log[(long)e * Nc + n]); h[n] = 0.f; }
  const long baseE = (long)b * Lc * Ec + e;
  const long baseN = (long)b * Lc * Nc;
  for (int t = 0; t < Lc; t++) {
    const float d  = dlt[baseE + (long)t * Ec];
    const float xv = xs[baseE + (long)t * Ec];
    const float dx = d * xv;
    const float* bp = Bt + baseN + (long)t * Nc;
    const float* cp = Ct + baseN + (long)t * Nc;
    float acc = 0.f;
#pragma unroll
    for (int n = 0; n < Nc; n++) {
      h[n] = __expf(d * Ar[n]) * h[n] + dx * bp[n];
      acc += h[n] * cp[n];
    }
    y[baseE + (long)t * Ec] = acc;
  }
}

// ---------------------------------------------------------------------------
// Gate: yg = (y + xs*W_D[e]) * silu(skip)  -> bf16 for final GEMM
// ---------------------------------------------------------------------------
__global__ __launch_bounds__(256)
void gate_kernel(const float* __restrict__ y, const float* __restrict__ xs,
                 const float* __restrict__ skip, const float* __restrict__ WD,
                 __bf16* __restrict__ yg_bf) {
  const long idx = (long)blockIdx.x * 256 + threadIdx.x;
  const int  e   = (int)(idx % Ec);
  const float yv = y[idx] + xs[idx] * WD[e];
  const float sk = skip[idx];
  const float g  = sk / (1.f + __expf(-sk));
  yg_bf[idx] = f2bf(yv * g);
}

// ---------------------------------------------------------------------------
// Host launch
// ---------------------------------------------------------------------------
extern "C" void kernel_launch(void* const* d_in, const int* in_sizes, int n_in,
                              void* d_out, int out_size, void* d_ws, size_t ws_size,
                              hipStream_t stream) {
  const float* resid  = (const float*)d_in[0];
  const float* norm_w = (const float*)d_in[1];
  const float* W_skip = (const float*)d_in[2];
  const float* W_in   = (const float*)d_in[3];
  const float* conv_w = (const float*)d_in[4];
  const float* conv_b = (const float*)d_in[5];
  const float* W_d1   = (const float*)d_in[6];
  const float* W_d2   = (const float*)d_in[7];
  const float* b_d2   = (const float*)d_in[8];
  const float* W_Bm   = (const float*)d_in[9];
  const float* W_Cm   = (const float*)d_in[10];
  const float* A_log  = (const float*)d_in[11];
  const float* W_D    = (const float*)d_in[12];
  const float* W_out  = (const float*)d_in[13];
  float* out = (float*)d_out;

  // ---- workspace carve-up ----
  char* cur = (char*)d_ws;
  auto alloc = [&](size_t bytes) -> void* {
    void* p = (void*)cur;
    cur += (bytes + 255) & ~(size_t)255;
    return p;
  };
  __bf16* xn_bf    = (__bf16*)alloc(BLc * Dc * 2);
  __bf16* wskip_bf = (__bf16*)alloc((long)Ec * Dc * 2);
  __bf16* win_bf   = (__bf16*)alloc((long)Ec * Dc * 2);
  __bf16* wout_bf  = (__bf16*)alloc((long)Dc * Ec * 2);
  __bf16* wd1_bf   = (__bf16*)alloc((long)Rc * Ec * 2);
  __bf16* wd2_bf   = (__bf16*)alloc((long)Ec * Rc * 2);
  __bf16* wbm_bf   = (__bf16*)alloc((long)Nc * Ec * 2);
  __bf16* wcm_bf   = (__bf16*)alloc((long)Nc * Ec * 2);
  float*  skip     = (float*)alloc(BLc * Ec * 4);
  float*  xin      = (float*)alloc(BLc * Ec * 4);
  float*  xs_f     = (float*)alloc(BLc * Ec * 4);
  __bf16* xs_bf    = (__bf16*)alloc(BLc * Ec * 2);
  float*  d1_f     = (float*)alloc(BLc * Rc * 4);
  __bf16* d1_bf    = (__bf16*)alloc(BLc * Rc * 2);
  float*  dlt      = (float*)alloc(BLc * Ec * 4);
  float*  Btb      = (float*)alloc(BLc * Nc * 4);
  float*  Ctb      = (float*)alloc(BLc * Nc * 4);
  float*  yb       = (float*)alloc(BLc * Ec * 4);
  __bf16* yg_bf    = (__bf16*)alloc(BLc * Ec * 2);

  auto cvt = [&](const float* src, __bf16* dst, long n) {
    cvt_f32_bf16_kernel<<<dim3((unsigned)((n + 255) / 256)), dim3(256), 0, stream>>>(src, dst, n);
  };

  // ---- 1. weights -> bf16 ----
  cvt(W_skip, wskip_bf, (long)Ec * Dc);
  cvt(W_in,   win_bf,   (long)Ec * Dc);
  cvt(W_out,  wout_bf,  (long)Dc * Ec);
  cvt(W_d1,   wd1_bf,   (long)Rc * Ec);
  cvt(W_d2,   wd2_bf,   (long)Ec * Rc);
  cvt(W_Bm,   wbm_bf,   (long)Nc * Ec);
  cvt(W_Cm,   wcm_bf,   (long)Nc * Ec);

  // ---- 2. RMSNorm ----
  rmsnorm_kernel<<<dim3((unsigned)BLc), dim3(256), 0, stream>>>(resid, norm_w, xn_bf);

  // ---- 3/4. skip & in projections (WMMA + TDM) ----
  {
    dim3 g((unsigned)(BLc / GT_M), (unsigned)((Ec + GT_N - 1) / GT_N));
    gemm_bf16_wmma<<<g, dim3(256), 0, stream>>>(xn_bf, wskip_bf, nullptr, skip,
                                                (int)BLc, Ec, Dc, Ec);
    gemm_bf16_wmma<<<g, dim3(256), 0, stream>>>(xn_bf, win_bf, nullptr, xin,
                                                (int)BLc, Ec, Dc, Ec);
  }

  // ---- 5. depthwise causal conv + SiLU ----
  conv_silu_kernel<<<dim3((unsigned)(BLc * Ec / 256)), dim3(256), 0, stream>>>(
      xin, conv_w, conv_b, xs_f, xs_bf);

  // ---- 6. delta low-rank GEMMs (WMMA + TDM) ----
  {
    dim3 g1((unsigned)(BLc / GT_M), (unsigned)((Rc + GT_N - 1) / GT_N));
    gemm_bf16_wmma<<<g1, dim3(256), 0, stream>>>(xs_bf, wd1_bf, nullptr, d1_f,
                                                 (int)BLc, Rc, Ec, Rc);
    cvt(d1_f, d1_bf, BLc * Rc);
    dim3 g2((unsigned)(BLc / GT_M), (unsigned)((Ec + GT_N - 1) / GT_N));
    gemm_bf16_wmma<<<g2, dim3(256), 0, stream>>>(d1_bf, wd2_bf, nullptr, dlt,
                                                 (int)BLc, Ec, Rc, Ec);
  }
  softplus_kernel<<<dim3((unsigned)(BLc * Ec / 256)), dim3(256), 0, stream>>>(dlt, b_d2);

  // ---- 7. B_t / C_t projections (WMMA + TDM, N=16 via descriptor OOB) ----
  {
    dim3 g((unsigned)(BLc / GT_M), 1);
    gemm_bf16_wmma<<<g, dim3(256), 0, stream>>>(xs_bf, wbm_bf, nullptr, Btb,
                                                (int)BLc, Nc, Ec, Nc);
    gemm_bf16_wmma<<<g, dim3(256), 0, stream>>>(xs_bf, wcm_bf, nullptr, Ctb,
                                                (int)BLc, Nc, Ec, Nc);
  }

  // ---- 8. selective scan ----
  scan_kernel<<<dim3((unsigned)(Ec / 256), (unsigned)Bc), dim3(256), 0, stream>>>(
      dlt, xs_f, Btb, Ctb, A_log, yb);

  // ---- 9. gate ----
  gate_kernel<<<dim3((unsigned)(BLc * Ec / 256)), dim3(256), 0, stream>>>(
      yb, xs_f, skip, W_D, yg_bf);

  // ---- 10. output projection + residual (WMMA + TDM, fused add) ----
  {
    dim3 g((unsigned)(BLc / GT_M), (unsigned)((Dc + GT_N - 1) / GT_N));
    gemm_bf16_wmma<<<g, dim3(256), 0, stream>>>(yg_bf, wout_bf, resid, out,
                                                (int)BLc, Dc, Ec, Dc);
  }
}
